// GraphLearning2_85607288143885
// MI455X (gfx1250) — compile-verified, matrix operand order checked by reading, not run on hardware
//
#include <hip/hip_runtime.h>
#include <hip/hip_bf16.h>
#include <math.h>

typedef __attribute__((ext_vector_type(16))) __bf16 v16bf;
typedef __attribute__((ext_vector_type(8)))  __bf16 v8bf;
typedef __attribute__((ext_vector_type(8)))  float  v8f;

// ---------------------------------------------------------------------------
// Problem constants (match reference)
// ---------------------------------------------------------------------------
static constexpr int NN = 100000;   // nodes
static constexpr int EE = 1600000;  // edges
static constexpr int WW = 100;      // feature dim
static constexpr int HH = 60;       // hidden dim
static constexpr int KP1 = 128;     // padded K for W=100
static constexpr int KP2 = 64;      // padded K for H=60
static constexpr int MT  = NN / 16; // 6250 M-tiles, exact

// ---------------------------------------------------------------------------
// init: acc = 0, deg = 1 (self loop)
// ---------------------------------------------------------------------------
__global__ void k_init(float* __restrict__ acc, float* __restrict__ deg,
                       int n, long long total) {
    long long i = (long long)blockIdx.x * blockDim.x + threadIdx.x;
    if (i >= total) return;
    acc[i] = 0.0f;
    if (i < n) deg[i] = 1.0f;
}

// degree accumulation over real edges (atomics resolve in L2)
__global__ void k_degree(const int* __restrict__ dst, float* __restrict__ deg, int e) {
    int i = blockIdx.x * blockDim.x + threadIdx.x;
    if (i >= e) return;
    atomicAdd(&deg[dst[i]], 1.0f);
}

// deg -> rsqrt(deg) in place (deg >= 1 always, self loop)
__global__ void k_rsqrt(float* __restrict__ deg, int n) {
    int i = blockIdx.x * blockDim.x + threadIdx.x;
    if (i >= n) return;
    deg[i] = rsqrtf(deg[i]);
}

// f32 [rows x cols] -> zero-padded bf16 [* x colsPad]
__global__ void k_cvt_pad(const float* __restrict__ src, __bf16* __restrict__ dst,
                          int rows, int cols, int colsPad, long long total) {
    long long i = (long long)blockIdx.x * blockDim.x + threadIdx.x;
    if (i >= total) return;
    int c = (int)(i % colsPad);
    long long r = i / colsPad;
    float v = 0.0f;
    if (r < rows && c < cols) v = src[r * cols + c];
    dst[i] = (__bf16)v;
}

// hin = bf16(relu(acc + gcn_b)), zero-padded to 128 cols
__global__ void k_build_hin(const float* __restrict__ acc, const float* __restrict__ gb,
                            __bf16* __restrict__ hin, long long total) {
    long long i = (long long)blockIdx.x * blockDim.x + threadIdx.x;
    if (i >= total) return;
    int k = (int)(i & 127);
    long long n = i >> 7;
    float v = 0.0f;
    if (k < WW) {
        v = acc[n * WW + k] + gb[k];
        v = v > 0.0f ? v : 0.0f;
    }
    hin[i] = (__bf16)v;
}

// gather xw[src], scale by dinv[src]*dinv[dst], scatter-add into acc[dst].
// One thread per (edge, float4 group); self loops appended as edges [E, E+N).
__global__ void k_scatter(const int* __restrict__ src, const int* __restrict__ dst,
                          const float* __restrict__ dinv, const float* __restrict__ xw,
                          float* __restrict__ acc, int e, int n) {
    int tid = blockIdx.x * blockDim.x + threadIdx.x;
    int total = (e + n) * 25;                 // 25 float4 groups per edge (W=100)
    if (tid >= total) return;
    int ed = tid / 25;
    int g  = tid % 25;
    int s, d;
    if (ed < e) { s = src[ed]; d = dst[ed]; }
    else        { s = d = ed - e; }
    float w = dinv[s] * dinv[d];
    float4 v = reinterpret_cast<const float4*>(xw)[(long long)s * 25 + g];
    float* a = acc + (long long)d * WW + g * 4;
    atomicAdd(a + 0, v.x * w);
    atomicAdd(a + 1, v.y * w);
    atomicAdd(a + 2, v.z * w);
    atomicAdd(a + 3, v.w * w);
}

// ---------------------------------------------------------------------------
// WMMA GEMM: C[M x Nvalid] = A_bf16[M x Kp] * B_bf16[Npad x Kp]^T
// One wave per 16x16 C tile, v_wmma_f32_16x16x32_bf16 over K.
// MODE 0: store f32               MODE 1: tanh(c+bias) -> bf16 (zero-pad cols)
// MODE 2: tanh(c+bias) -> f32
// Fragment layouts per CDNA5 ISA 7.12.2 (16-bit A 16x32 / 32-bit C 16x16).
// ---------------------------------------------------------------------------
template <int MODE>
__global__ void k_wmma_gemm(const __bf16* __restrict__ A, const __bf16* __restrict__ B,
                            const float* __restrict__ bias,
                            float* __restrict__ Cf, __bf16* __restrict__ Cbf,
                            int mtiles, int Kp, int Nvalid, int ldc) {
    const int wave = threadIdx.x >> 5;
    const int lane = threadIdx.x & 31;
    const int mt = blockIdx.x * 4 + wave;
    if (mt >= mtiles) return;                      // wave-uniform, EXEC stays full
    const int m0 = mt * 16;
    const int n0 = blockIdx.y * 16;
    const int half = lane >> 4;                    // 0: K {0..7,16..23}, 1: {8..15,24..31}
    const int l16  = lane & 15;
    const int klo  = half * 8;

    const __bf16* arow = A + (long long)(m0 + l16) * Kp;
    const __bf16* brow = B + (long long)(n0 + l16) * Kp;

    v8f c = {};
    for (int kb = 0; kb < Kp; kb += 32) {
        v8bf a0 = *reinterpret_cast<const v8bf*>(arow + kb + klo);
        v8bf a1 = *reinterpret_cast<const v8bf*>(arow + kb + klo + 16);
        v8bf b0 = *reinterpret_cast<const v8bf*>(brow + kb + klo);
        v8bf b1 = *reinterpret_cast<const v8bf*>(brow + kb + klo + 16);
        v16bf av = __builtin_shufflevector(a0, a1, 0,1,2,3,4,5,6,7,8,9,10,11,12,13,14,15);
        v16bf bv = __builtin_shufflevector(b0, b1, 0,1,2,3,4,5,6,7,8,9,10,11,12,13,14,15);
        c = __builtin_amdgcn_wmma_f32_16x16x32_bf16(
                /*neg_a=*/false, av, /*neg_b=*/false, bv,
                /*c_mod=*/(short)0, c, /*reuse_a=*/false, /*reuse_b=*/false);
    }

    const int n = n0 + l16;
#pragma unroll
    for (int r = 0; r < 8; ++r) {
        int m = m0 + half * 8 + r;                 // C layout: VGPR r -> M = r (+8 for hi lanes)
        float v = c[r];
        if (MODE == 0) {
            if (n < Nvalid) Cf[(long long)m * ldc + n] = v;
        } else if (MODE == 1) {
            float bv2 = (n < Nvalid) ? bias[n] : 0.0f;
            float o   = (n < Nvalid) ? tanhf(v + bv2) : 0.0f;   // zero-pads cols 60..63
            Cbf[(long long)m * ldc + n] = (__bf16)o;
        } else {
            if (n < Nvalid) Cf[(long long)m * ldc + n] = tanhf(v + bias[n]);
        }
    }
}

// ---------------------------------------------------------------------------
extern "C" void kernel_launch(void* const* d_in, const int* in_sizes, int n_in,
                              void* d_out, int out_size, void* d_ws, size_t ws_size,
                              hipStream_t stream) {
    const float* x     = (const float*)d_in[0];
    const int*   srcI  = (const int*)d_in[1];          // [2,E]: row 0 = src
    const int*   dstI  = srcI + EE;                    //        row 1 = dst
    const float* gcn_w = (const float*)d_in[2];
    const float* gcn_b = (const float*)d_in[3];
    const float* w1    = (const float*)d_in[4];
    const float* b1    = (const float*)d_in[5];
    const float* w2    = (const float*)d_in[6];
    const float* b2    = (const float*)d_in[7];
    float* out = (float*)d_out;

    // scratch carving (256B aligned)
    char* ws = (char*)d_ws;
    size_t off = 0;
    auto carve = [&](size_t bytes) -> char* {
        char* p = ws + off;
        off = (off + bytes + 255) & ~(size_t)255;
        return p;
    };
    float*  deg  = (float*)carve((size_t)NN * 4);              // -> dinv in place
    float*  xw   = (float*)carve((size_t)NN * WW * 4);
    float*  acc  = (float*)carve((size_t)NN * WW * 4);
    __bf16* xbf  = (__bf16*)carve((size_t)NN * KP1 * 2);       // reused as hin
    __bf16* h1bf = (__bf16*)carve((size_t)NN * KP2 * 2);
    __bf16* gwbf = (__bf16*)carve((size_t)112 * KP1 * 2);
    __bf16* w1bf = (__bf16*)carve((size_t)64  * KP1 * 2);
    __bf16* w2bf = (__bf16*)carve((size_t)112 * KP2 * 2);

    const int B = 256;
    long long tAcc = (long long)NN * WW;
    long long tX   = (long long)NN * KP1;

    // 1) init acc=0, deg=1 (self loops)
    k_init<<<(int)((tAcc + B - 1) / B), B, 0, stream>>>(acc, deg, NN, tAcc);
    // 2) in-degree atomics + 3) dinv = rsqrt(deg)
    k_degree<<<(EE + B - 1) / B, B, 0, stream>>>(dstI, deg, EE);
    k_rsqrt<<<(NN + B - 1) / B, B, 0, stream>>>(deg, NN);
    // 4) bf16 conversions (zero padded)
    k_cvt_pad<<<(int)((tX + B - 1) / B), B, 0, stream>>>(x, xbf, NN, WW, KP1, tX);
    k_cvt_pad<<<(112 * KP1 + B - 1) / B, B, 0, stream>>>(gcn_w, gwbf, WW, WW, KP1, 112 * KP1);
    k_cvt_pad<<<(64  * KP1 + B - 1) / B, B, 0, stream>>>(w1,    w1bf, HH, WW, KP1, 64 * KP1);
    k_cvt_pad<<<(112 * KP2 + B - 1) / B, B, 0, stream>>>(w2,    w2bf, WW, HH, KP2, 112 * KP2);
    // 5) xw = x @ gcn_w^T  (WMMA bf16 -> f32)
    dim3 g1((MT + 3) / 4, (WW + 15) / 16);  // 1563 x 7
    k_wmma_gemm<0><<<g1, 128, 0, stream>>>(xbf, gwbf, nullptr, xw, nullptr, MT, KP1, WW, WW);
    // 6) normalized gather/scatter (L2-resident atomics), incl. self loops
    int totSc = (EE + NN) * 25;
    k_scatter<<<(totSc + B - 1) / B, B, 0, stream>>>(srcI, dstI, deg, xw, acc, EE, NN);
    // 7) hin = bf16(relu(acc + gcn_b)), padded to 128 (reuses xbf)
    k_build_hin<<<(int)((tX + B - 1) / B), B, 0, stream>>>(acc, gcn_b, xbf, tX);
    // 8) h1 = tanh(hin @ w1^T + b1) -> bf16 padded to 64
    dim3 g2((MT + 3) / 4, (HH + 15) / 16);  // 1563 x 4
    k_wmma_gemm<1><<<g2, 128, 0, stream>>>(xbf, w1bf, b1, nullptr, h1bf, MT, KP1, HH, KP2);
    // 9) out = tanh(h1 @ w2^T + b2) -> f32 d_out
    dim3 g3((MT + 3) / 4, (WW + 15) / 16);  // 1563 x 7
    k_wmma_gemm<2><<<g3, 128, 0, stream>>>(h1bf, w2bf, b2, out, nullptr, MT, KP2, WW, WW);
}
